// W8A8LinearStatic_64699387347307
// MI455X (gfx1250) — 8-bit matrix operand order repaired and checked by reading, compiled, not run on hardware
//
#include <hip/hip_runtime.h>
#include <stdint.h>

typedef int v8i __attribute__((ext_vector_type(8)));
typedef int v4i __attribute__((ext_vector_type(4)));
typedef int v2i __attribute__((ext_vector_type(2)));

typedef __attribute__((address_space(1))) v4i* gas_v4i_ptr;
typedef __attribute__((address_space(3))) v4i* lds_v4i_ptr;

#define TOKENS 8192
#define D_IN   4096
#define D_OUT  11008

#define BM 128
#define BN 128
#define BK 64
#define LDS_PITCH 80
#define NK (D_IN / BK)

#if defined(__has_builtin)
#if __has_builtin(__builtin_amdgcn_global_load_async_to_lds_b128) && \
    __has_builtin(__builtin_amdgcn_s_wait_asynccnt)
#define USE_ASYNC_LDS 1
#endif
#endif
#ifndef USE_ASYNC_LDS
#define USE_ASYNC_LDS 0
#endif

#if USE_ASYNC_LDS
#define ASYNC_CP_B128(dstLds, srcG)                                          \
    __builtin_amdgcn_global_load_async_to_lds_b128(                          \
        (gas_v4i_ptr)(srcG),                                                 \
        (lds_v4i_ptr)(dstLds),                                               \
        0, 0)
#endif

__global__ __launch_bounds__(256) void w8a8_quant_kernel(
    const float* __restrict__ x,
    const float* __restrict__ in_scale,
    const int*   __restrict__ in_off,
    int*         __restrict__ xq_packed)
{
    const float inv = 1.0f / in_scale[0];
    const int   off = in_off[0];
    const int i = blockIdx.x * 256 + threadIdx.x;
    const float4 v = ((const float4*)x)[i];
    int q0 = min(127, max(-128, __float2int_rn(v.x * inv) + off));
    int q1 = min(127, max(-128, __float2int_rn(v.y * inv) + off));
    int q2 = min(127, max(-128, __float2int_rn(v.z * inv) + off));
    int q3 = min(127, max(-128, __float2int_rn(v.w * inv) + off));
    xq_packed[i] = (q0 & 0xFF) | ((q1 & 0xFF) << 8) | ((q2 & 0xFF) << 16) | (q3 << 24);
}

__global__ __launch_bounds__(256) void w8a8_gemm_kernel(
    const int8_t* __restrict__ xq,
    const int8_t* __restrict__ w,
    const float*  __restrict__ deq,
    const int*    __restrict__ qbias,
    float*        __restrict__ out)
{
    __shared__ __align__(16) int8_t sA[2][BM * LDS_PITCH];
    __shared__ __align__(16) int8_t sB[2][BN * LDS_PITCH];

    const int tid  = threadIdx.x;
    const int lane = tid & 31;
    const int wave = tid >> 5;
    const int wm   = wave & 3;
    const int wn   = wave >> 2;
    const int half = lane >> 4;
    const int l15  = lane & 15;

    const int rowBase = blockIdx.y * BM;
    const int colBase = blockIdx.x * BN;

    const int id0 = tid * 2;
    const int r0  = id0 >> 2, c0 = (id0 & 3) * 16;
    const int id1 = id0 + 1;
    const int r1  = id1 >> 2, c1 = (id1 & 3) * 16;

    const int8_t* gA = xq + (size_t)rowBase * D_IN;
    const int8_t* gB = w  + (size_t)colBase * D_IN;

#if USE_ASYNC_LDS
    ASYNC_CP_B128(&sA[0][r0 * LDS_PITCH + c0], gA + (size_t)r0 * D_IN + c0);
    ASYNC_CP_B128(&sA[0][r1 * LDS_PITCH + c1], gA + (size_t)r1 * D_IN + c1);
    ASYNC_CP_B128(&sB[0][r0 * LDS_PITCH + c0], gB + (size_t)r0 * D_IN + c0);
    ASYNC_CP_B128(&sB[0][r1 * LDS_PITCH + c1], gB + (size_t)r1 * D_IN + c1);
    __builtin_amdgcn_s_wait_asynccnt(0);
    __syncthreads();
#else
    v4i pa0, pa1, pb0, pb1;
    pa0 = *(const v4i*)(gA + (size_t)r0 * D_IN + c0);
    pa1 = *(const v4i*)(gA + (size_t)r1 * D_IN + c1);
    pb0 = *(const v4i*)(gB + (size_t)r0 * D_IN + c0);
    pb1 = *(const v4i*)(gB + (size_t)r1 * D_IN + c1);
    *(v4i*)(&sA[0][r0 * LDS_PITCH + c0]) = pa0;
    *(v4i*)(&sA[0][r1 * LDS_PITCH + c1]) = pa1;
    *(v4i*)(&sB[0][r0 * LDS_PITCH + c0]) = pb0;
    *(v4i*)(&sB[0][r1 * LDS_PITCH + c1]) = pb1;
    __syncthreads();
#endif

    const v8i zero8 = {0, 0, 0, 0, 0, 0, 0, 0};
    v8i acc[2][4];
#pragma unroll
    for (int tm = 0; tm < 2; ++tm)
#pragma unroll
        for (int tn = 0; tn < 4; ++tn) acc[tm][tn] = zero8;

    for (int kt = 0; kt < NK; ++kt) {
        const int cur = kt & 1;
        const int nxt = cur ^ 1;

        if (kt + 1 < NK) {
            const int k0 = (kt + 1) * BK;
#if USE_ASYNC_LDS
            ASYNC_CP_B128(&sA[nxt][r0 * LDS_PITCH + c0], gA + (size_t)r0 * D_IN + k0 + c0);
            ASYNC_CP_B128(&sA[nxt][r1 * LDS_PITCH + c1], gA + (size_t)r1 * D_IN + k0 + c1);
            ASYNC_CP_B128(&sB[nxt][r0 * LDS_PITCH + c0], gB + (size_t)r0 * D_IN + k0 + c0);
            ASYNC_CP_B128(&sB[nxt][r1 * LDS_PITCH + c1], gB + (size_t)r1 * D_IN + k0 + c1);
#else
            pa0 = *(const v4i*)(gA + (size_t)r0 * D_IN + k0 + c0);
            pa1 = *(const v4i*)(gA + (size_t)r1 * D_IN + k0 + c1);
            pb0 = *(const v4i*)(gB + (size_t)r0 * D_IN + k0 + c0);
            pb1 = *(const v4i*)(gB + (size_t)r1 * D_IN + k0 + c1);
#endif
        }

        v8i afrag[2];
#pragma unroll
        for (int tm = 0; tm < 2; ++tm) {
            const int8_t* p = &sA[cur][(wm * 32 + tm * 16 + l15) * LDS_PITCH + half * 16];
            union { v8i v; v2i d[4]; } u;
            u.d[0] = *(const v2i*)(p + 0);
            u.d[1] = *(const v2i*)(p + 8);
            u.d[2] = *(const v2i*)(p + 32);
            u.d[3] = *(const v2i*)(p + 40);
            afrag[tm] = u.v;
        }

        v8i bfrag[4];
#pragma unroll
        for (int tn = 0; tn < 4; ++tn) {
            const int8_t* p = &sB[cur][(wn * 64 + tn * 16 + l15) * LDS_PITCH + half * 16];
            union { v8i v; v4i q[2]; } u;
            u.q[0] = *(const v4i*)(p + 0);
            u.q[1] = *(const v4i*)(p + 32);
            bfrag[tn] = u.v;
        }

#pragma unroll
        for (int tm = 0; tm < 2; ++tm)
#pragma unroll
            for (int tn = 0; tn < 4; ++tn)
                acc[tm][tn] = __builtin_amdgcn_wmma_i32_16x16x64_iu8(
                     true, afrag[tm],
                     true, bfrag[tn],
                    acc[tm][tn],
                     false,  false);

        if (kt + 1 < NK) {
#if USE_ASYNC_LDS
            __builtin_amdgcn_s_wait_asynccnt(0);
            __syncthreads();
#else
            *(v4i*)(&sA[nxt][r0 * LDS_PITCH + c0]) = pa0;
            *(v4i*)(&sA[nxt][r1 * LDS_PITCH + c1]) = pa1;
            *(v4i*)(&sB[nxt][r0 * LDS_PITCH + c0]) = pb0;
            *(v4i*)(&sB[nxt][r1 * LDS_PITCH + c1]) = pb1;
            __syncthreads();
#endif
        }
    }

#pragma unroll
    for (int tn = 0; tn < 4; ++tn) {
        const int n   = colBase + wn * 64 + tn * 16 + l15;
        const float sc = deq[n];
        const int   bs = qbias[n];
#pragma unroll
        for (int tm = 0; tm < 2; ++tm) {
            const int mBase = rowBase + wm * 32 + tm * 16 + half * 8;
#pragma unroll
            for (int r = 0; r < 8; ++r) {
                out[(size_t)(mBase + r) * D_OUT + n] =
                    (float)(acc[tm][tn][r] + bs) * sc;
            }
        }
    }
}

extern "C" void kernel_launch(void* const* d_in, const int* in_sizes, int n_in,
                              void* d_out, int out_size, void* d_ws, size_t ws_size,
                              hipStream_t stream) {
    const float*  x        = (const float*)d_in[0];
    const int8_t* weight   = (const int8_t*)d_in[1];
    const float*  in_scale = (const float*)d_in[2];
    const int*    in_off   = (const int*)d_in[3];
    const float*  deq      = (const float*)d_in[4];
    const int*    qbias    = (const int*)d_in[5];
    float*        out      = (float*)d_out;

    int8_t* xq = (int8_t*)d_ws;

    const int quant_blocks = (TOKENS * D_IN / 4) / 256;
    w8a8_quant_kernel<<<quant_blocks, 256, 0, stream>>>(x, in_scale, in_off, (int*)xq);

    dim3 grid(D_OUT / BN, TOKENS / BM);
    w8a8_gemm_kernel<<<grid, 256, 0, stream>>>(xq, weight, deq, qbias, out);
}
